// ST_TruncateFormer_23278722744418
// MI455X (gfx1250) — matrix operand order repaired and checked by reading, not compiled, not run on hardware
//
#include <hip/hip_runtime.h>
#include <hip/hip_bf16.h>
#include <math.h>

// ---------------------------------------------------------------------------
// Problem constants (from reference): B=16, C=512, D=256, K=16, RD=128, H=8,
// FFN=1024.  Forward-only: z == z_hard  =>  rec_x is a gather of x rows, and
// k/v projections hoist to a single per-token QKV GEMM.
// ---------------------------------------------------------------------------
#define TB 16
#define TC 512
#define TD 256
#define TK 16
#define TRD 128
#define TH 8
#define THD 32
#define TFFN 1024
#define ROWS (TB * TC) // 8192 token rows

typedef __attribute__((ext_vector_type(16))) _Float16 v16h;
typedef __attribute__((ext_vector_type(8)))  float    v8f;
typedef __attribute__((ext_vector_type(4)))  float    v4f;

#define AS_STRIDE 36  // pad LDS rows (f16) to rotate banks; 72 B, 8 B aligned

union PackH4 { _Float16 h[4]; unsigned long long u; };

// ---------------------------------------------------------------------------
// Generic tiled GEMM:  out[M,N] = act( A[M,K] @ W[N,K]^T + bias ) (+ resid)
// Block = 256 threads (8 waves).  Tile: 128 rows x 64 cols, k-step 32.
// Wave w computes rows 16w..16w+15 over all 64 cols: 4x v_wmma_f32_16x16x32_f16.
// Software-pipelined: next tile's b128 global loads are issued into registers
// while the current tile's WMMAs run; fp32->f16 packs go to LDS as b64 stores.
// Requires M%128==0, N%64==0, K%32==0 (true for every call here).
// act: 0 = bias only, 1 = bias+GELU(exact), 2 = bias+residual.
// ---------------------------------------------------------------------------
__global__ __launch_bounds__(256) void gemm_wmma_f16(
    const float* __restrict__ A, const float* __restrict__ W,
    const float* __restrict__ bias, const float* __restrict__ resid,
    float* __restrict__ out, int M, int N, int Kdim, int act)
{
    __shared__ _Float16 As[128][AS_STRIDE];
    __shared__ _Float16 Bs[64][AS_STRIDE];

    const int tid  = threadIdx.x;
    const int wave = tid >> 5;
    const int lane = tid & 31;
    const int m    = lane & 15;   // row within 16x16 tile (A) / col (B,D)
    const int kh   = lane >> 4;   // K-half selector

    const int row0 = blockIdx.y * 128;
    const int col0 = blockIdx.x * 64;

    // Staging thread layout: 8 float4's span one 32-wide K slab.
    const int c4 = tid & 7;   // float4 column
    const int rs = tid >> 3;  // base row 0..31

    v4f aReg[4];
    v4f bReg[2];

    auto loadA = [&](int k0) {
        #pragma unroll
        for (int i = 0; i < 4; ++i)
            aReg[i] = *(const v4f*)(A + (size_t)(row0 + rs + i * 32) * Kdim + k0 + c4 * 4);
    };
    auto loadB = [&](int k0) {
        #pragma unroll
        for (int i = 0; i < 2; ++i)
            bReg[i] = *(const v4f*)(W + (size_t)(col0 + rs + i * 32) * Kdim + k0 + c4 * 4);
    };

    v8f acc[4] = {};

    loadA(0);
    loadB(0);

    for (int k0 = 0; k0 < Kdim; k0 += 32) {
        // Pack staged registers into LDS (one ds_store_b64 per float4).
        #pragma unroll
        for (int i = 0; i < 4; ++i) {
            PackH4 pk;
            #pragma unroll
            for (int j = 0; j < 4; ++j) pk.h[j] = (_Float16)aReg[i][j];
            *(unsigned long long*)&As[rs + i * 32][c4 * 4] = pk.u;
        }
        #pragma unroll
        for (int i = 0; i < 2; ++i) {
            PackH4 pk;
            #pragma unroll
            for (int j = 0; j < 4; ++j) pk.h[j] = (_Float16)bReg[i][j];
            *(unsigned long long*)&Bs[rs + i * 32][c4 * 4] = pk.u;
        }
        __syncthreads();

        // Issue next tile's global loads now; they overlap the WMMAs below.
        if (k0 + 32 < Kdim) {
            loadA(k0 + 32);
            loadB(k0 + 32);
            if (k0 + 64 < Kdim)
                __builtin_prefetch(W + (size_t)(col0 + rs) * Kdim + k0 + 64, 0, 1);
        }

        // A fragment (16-bit A 16x32 layout): lane = M, K-halves split at lane 16.
        union { v16h v; _Float16 h[16]; } af;
        #pragma unroll
        for (int e = 0; e < 8; ++e) af.h[e]     = As[wave * 16 + m][kh * 8 + e];
        #pragma unroll
        for (int e = 0; e < 8; ++e) af.h[8 + e] = As[wave * 16 + m][16 + kh * 8 + e];

        #pragma unroll
        for (int t = 0; t < 4; ++t) {
            // B fragment: lane = N (column of B = row of W), K-halves split like A.
            union { v16h v; _Float16 h[16]; } bf;
            #pragma unroll
            for (int e = 0; e < 8; ++e) { bf.h[e] = Bs[t * 16 + m][kh * 8 + e]; bf.h[8 + e] = Bs[t * 16 + m][16 + kh * 8 + e]; }
            acc[t] = __builtin_amdgcn_wmma_f32_16x16x32_f16(
                false, af.v, false, bf.v, (short)0, acc[t], false, false);
        }
        __syncthreads();
    }

    // Epilogue. D layout: VGPR r -> row (r + 8*kh), lane%16 -> col.
    #pragma unroll
    for (int t = 0; t < 4; ++t) {
        int col  = col0 + t * 16 + m;
        float bv = bias ? bias[col] : 0.0f;
        #pragma unroll
        for (int r = 0; r < 8; ++r) {
            int row = row0 + wave * 16 + kh * 8 + r;
            float v = acc[t][r] + bv;
            if (act == 1) {
                v = 0.5f * v * (1.0f + erff(v * 0.70710678118654752f)); // exact GELU
            } else if (act == 2) {
                v += resid[(size_t)row * N + col];
            }
            out[(size_t)row * N + col] = v;
        }
    }
}

// ---------------------------------------------------------------------------
// Logits + top-16: one wave per (batch, 16-row q-tile).
// logits[16,512] = xq_tile @ xk^T computed by WMMA (TEMP is monotonic: skip),
// then 16 rounds of wave-wide argmax per row (exact top-k, ties -> lower idx).
// ---------------------------------------------------------------------------
#define QS_STRIDE (TRD + 8)   // 136 halves = 272 B: 4-bank rotation per row

__global__ __launch_bounds__(32) void logits_topk(
    const float* __restrict__ xq, const float* __restrict__ xk,
    int* __restrict__ idx_out)
{
    __shared__ float    lg[16][TC];           // 32 KB
    __shared__ _Float16 qs[16][QS_STRIDE];    // ~4.25 KB
    __shared__ _Float16 ks[16][QS_STRIDE];    // ~4.25 KB

    const int lane = threadIdx.x;
    const int b    = blockIdx.y;
    const int q0   = blockIdx.x * 16;
    const int m    = lane & 15;
    const int kh   = lane >> 4;

    const float* xqb = xq + ((size_t)b * TC + q0) * TRD;
    #pragma unroll
    for (int i = 0; i < 16; ++i) {
        int e4 = lane + i * 32;               // float4 index 0..511
        v4f v = *(const v4f*)(xqb + e4 * 4);
        PackH4 pk;
        #pragma unroll
        for (int j = 0; j < 4; ++j) pk.h[j] = (_Float16)v[j];
        *(unsigned long long*)&qs[e4 >> 5][(e4 & 31) * 4] = pk.u;
    }
    __syncthreads();

    for (int ct = 0; ct < TC / 16; ++ct) {
        const float* xkb = xk + ((size_t)b * TC + ct * 16) * TRD;
        #pragma unroll
        for (int i = 0; i < 16; ++i) {
            int e4 = lane + i * 32;
            v4f v = *(const v4f*)(xkb + e4 * 4);
            PackH4 pk;
            #pragma unroll
            for (int j = 0; j < 4; ++j) pk.h[j] = (_Float16)v[j];
            *(unsigned long long*)&ks[e4 >> 5][(e4 & 31) * 4] = pk.u;
        }
        __syncthreads();

        v8f acc = {};
        #pragma unroll
        for (int kk = 0; kk < TRD / 32; ++kk) {
            union { v16h v; _Float16 h[16]; } af, bf;
            #pragma unroll
            for (int e = 0; e < 8; ++e) af.h[e]     = qs[m][kk * 32 + kh * 8 + e];
            #pragma unroll
            for (int e = 0; e < 8; ++e) af.h[8 + e] = qs[m][kk * 32 + 16 + kh * 8 + e];
            #pragma unroll
            for (int e = 0; e < 8; ++e) { bf.h[e] = ks[m][kk * 32 + kh * 8 + e]; bf.h[8 + e] = ks[m][kk * 32 + 16 + kh * 8 + e]; }
            acc = __builtin_amdgcn_wmma_f32_16x16x32_f16(
                false, af.v, false, bf.v, (short)0, acc, false, false);
        }
        #pragma unroll
        for (int r = 0; r < 8; ++r) lg[r + 8 * kh][ct * 16 + m] = acc[r];
        __syncthreads();
    }

    // Exact top-16 per row via iterative wave argmax.
    for (int r = 0; r < 16; ++r) {
        for (int it = 0; it < TK; ++it) {
            float best = -INFINITY;
            int   bi   = lane;
            #pragma unroll
            for (int j = 0; j < TC / 32; ++j) {
                int c = lane + j * 32;
                float v = lg[r][c];
                if (v > best) { best = v; bi = c; }
            }
            #pragma unroll
            for (int off = 16; off >= 1; off >>= 1) {
                float ov = __shfl_xor(best, off, 32);
                int   oi = __shfl_xor(bi,   off, 32);
                if (ov > best || (ov == best && oi < bi)) { best = ov; bi = oi; }
            }
            if (lane == 0)
                idx_out[((size_t)b * TC + q0 + r) * TK + it] = bi;
            if ((bi & 31) == lane)       // owner lane retires the winner
                lg[r][bi] = -INFINITY;
            __syncthreads();
        }
    }
}

// ---------------------------------------------------------------------------
// Gather attention: one block (8 waves) per token.
// The 16 gathered K/V rows are contiguous 512-float spans of the QKV buffer:
// stage them into LDS with gfx1250 async global->LDS b128 copies (ASYNCcnt),
// then each wave (= head) computes scores/softmax/ctx from LDS.
// ---------------------------------------------------------------------------
__global__ __launch_bounds__(256) void attn_gather(
    const float* __restrict__ qkv, const int* __restrict__ idx,
    float* __restrict__ ctx)
{
    __shared__ float kvs[TK][2 * TD];   // [16][512]: K row (256) then V row (256) = 32 KB

    const int bq   = blockIdx.x;        // token index 0..8191
    const int b    = bq >> 9;           // /C
    const int tid  = threadIdx.x;
    const int head = tid >> 5;
    const int lane = tid & 31;
    const int* ix  = idx + (size_t)bq * TK;

    // Async-copy the 16 gathered K/V rows (2 KB each) straight into LDS.
    // 2048 float4 transfers / 256 threads = 8 per thread.
    #pragma unroll
    for (int i = 0; i < 8; ++i) {
        int flat = tid + i * 256;       // float4 index 0..2047
        int key  = flat >> 7;           // 0..15
        int e4   = flat & 127;          // float4 within the 512-float row
        int c    = ix[key];
        const float* src = qkv + ((size_t)b * TC + c) * (3 * TD) + TD + e4 * 4;
        unsigned long long ga = (unsigned long long)(size_t)src;
        unsigned dst = (unsigned)(size_t)&kvs[key][e4 * 4]; // LDS byte offset
        asm volatile("global_load_async_to_lds_b128 %0, %1, off"
                     :: "v"(dst), "v"(ga) : "memory");
    }
    asm volatile("s_wait_asynccnt 0x0" ::: "memory");
    __syncthreads();

    const float qv = qkv[(size_t)bq * (3 * TD) + head * THD + lane]
                   * 0.17677669529663689f; // 1/sqrt(hd)

    float sc[TK];
    #pragma unroll
    for (int ki = 0; ki < TK; ++ki) {
        float p = qv * kvs[ki][head * THD + lane];
        #pragma unroll
        for (int off = 16; off >= 1; off >>= 1) p += __shfl_xor(p, off, 32);
        sc[ki] = p;
    }
    float mx = sc[0];
    #pragma unroll
    for (int ki = 1; ki < TK; ++ki) mx = fmaxf(mx, sc[ki]);
    float se = 0.0f;
    #pragma unroll
    for (int ki = 0; ki < TK; ++ki) { sc[ki] = __expf(sc[ki] - mx); se += sc[ki]; }
    const float inv = 1.0f / se;

    float o = 0.0f;
    #pragma unroll
    for (int ki = 0; ki < TK; ++ki)
        o += sc[ki] * inv * kvs[ki][TD + head * THD + lane];

    ctx[(size_t)bq * TD + head * THD + lane] = o;
}

// ---------------------------------------------------------------------------
// Row LayerNorm over D=256: one block per row.
// ---------------------------------------------------------------------------
__global__ __launch_bounds__(256) void layernorm_row(
    const float* __restrict__ in, const float* __restrict__ g,
    const float* __restrict__ bta, float* __restrict__ out)
{
    __shared__ float red[256];
    const int row = blockIdx.x;
    const int t   = threadIdx.x;
    float v = in[(size_t)row * TD + t];

    red[t] = v; __syncthreads();
    for (int s = 128; s > 0; s >>= 1) { if (t < s) red[t] += red[t + s]; __syncthreads(); }
    float mean = red[0] * (1.0f / TD);
    __syncthreads();

    float d = v - mean;
    red[t] = d * d; __syncthreads();
    for (int s = 128; s > 0; s >>= 1) { if (t < s) red[t] += red[t + s]; __syncthreads(); }
    float var = red[0] * (1.0f / TD);

    float r = rsqrtf(var + 1e-5f);
    out[(size_t)row * TD + t] = d * r * g[t] + bta[t];
}

// ---------------------------------------------------------------------------
// Workspace layout (floats).  The xq/xk/qkv region (exactly 8,388,608 floats)
// is reused for the FFN hidden activation after attention is done.
// ---------------------------------------------------------------------------
#define OFF_XQ   0
#define OFF_XK   (OFF_XQ + ROWS * TRD)          // 1,048,576
#define OFF_QKV  (OFF_XK + ROWS * TRD)          // 2,097,152
#define OFF_CTX  (OFF_QKV + ROWS * 3 * TD)      // 8,388,608
#define OFF_PRE  (OFF_CTX + ROWS * TD)          // 10,485,760
#define OFF_X1   (OFF_PRE + ROWS * TD)          // 12,582,912
#define OFF_IDX  (OFF_X1 + ROWS * TD)           // 14,680,064 (ints live here)
#define OFF_H    OFF_XQ                          // reuse: 8192x1024 hidden

extern "C" void kernel_launch(void* const* d_in, const int* in_sizes, int n_in,
                              void* d_out, int out_size, void* d_ws, size_t ws_size,
                              hipStream_t stream) {
    (void)in_sizes; (void)n_in; (void)out_size; (void)ws_size;
    const float* x         = (const float*)d_in[0];
    const float* wq_rec    = (const float*)d_in[1];
    const float* wk_rec    = (const float*)d_in[2];
    const float* in_proj_w = (const float*)d_in[3];
    const float* in_proj_b = (const float*)d_in[4];
    const float* out_w     = (const float*)d_in[5];
    const float* out_b     = (const float*)d_in[6];
    const float* ln1_g     = (const float*)d_in[7];
    const float* ln1_b     = (const float*)d_in[8];
    const float* mlp_w1    = (const float*)d_in[9];
    const float* mlp_b1    = (const float*)d_in[10];
    const float* mlp_w2    = (const float*)d_in[11];
    const float* mlp_b2    = (const float*)d_in[12];
    const float* ln2_g     = (const float*)d_in[13];
    const float* ln2_b     = (const float*)d_in[14];
    float* outp = (float*)d_out;

    float* wsf  = (float*)d_ws;
    float* xq   = wsf + OFF_XQ;
    float* xk   = wsf + OFF_XK;
    float* qkv  = wsf + OFF_QKV;
    float* ctx  = wsf + OFF_CTX;
    float* pre  = wsf + OFF_PRE;
    float* x1   = wsf + OFF_X1;
    float* hbuf = wsf + OFF_H;
    int*   idx  = (int*)(wsf + OFF_IDX);

    const dim3 blk(256);

    // 1-2) Recurrence projections: xq = x@wq_rec^T, xk = x@wk_rec^T  (8192x128x256)
    gemm_wmma_f16<<<dim3(TRD / 64, ROWS / 128), blk, 0, stream>>>(
        x, wq_rec, nullptr, nullptr, xq, ROWS, TRD, TD, 0);
    gemm_wmma_f16<<<dim3(TRD / 64, ROWS / 128), blk, 0, stream>>>(
        x, wk_rec, nullptr, nullptr, xk, ROWS, TRD, TD, 0);

    // 3) Per-batch logits (512x512x128 via WMMA) + exact top-16 indices.
    logits_topk<<<dim3(TC / 16, TB), dim3(32), 0, stream>>>(xq, xk, idx);

    // 4) Fused QKV projection: qkv = x@in_proj_w^T + b  (8192x768x256)
    gemm_wmma_f16<<<dim3(3 * TD / 64, ROWS / 128), blk, 0, stream>>>(
        x, in_proj_w, in_proj_b, nullptr, qkv, ROWS, 3 * TD, TD, 0);

    // 5) Gather-attention over top-K tokens (rec_x never materialized);
    //    K/V rows staged via async global->LDS copies.
    attn_gather<<<dim3(ROWS), blk, 0, stream>>>(qkv, idx, ctx);

    // 6) Output projection + residual, then LN1.
    gemm_wmma_f16<<<dim3(TD / 64, ROWS / 128), blk, 0, stream>>>(
        ctx, out_w, out_b, x, pre, ROWS, TD, TD, 2);
    layernorm_row<<<dim3(ROWS), blk, 0, stream>>>(pre, ln1_g, ln1_b, x1);

    // 7) MLP up + exact GELU (8192x1024x256).
    gemm_wmma_f16<<<dim3(TFFN / 64, ROWS / 128), blk, 0, stream>>>(
        x1, mlp_w1, mlp_b1, nullptr, hbuf, ROWS, TFFN, TD, 1);

    // 8) MLP down + residual (8192x256x1024), then LN2 -> output.
    gemm_wmma_f16<<<dim3(TD / 64, ROWS / 128), blk, 0, stream>>>(
        hbuf, mlp_w2, mlp_b2, x1, pre, ROWS, TD, TFFN, 2);
    layernorm_row<<<dim3(ROWS), blk, 0, stream>>>(pre, ln2_g, ln2_b, outp);
}
